// GraphDecoderHet_62981400429115
// MI455X (gfx1250) — matrix operand order repaired and checked by reading, not compiled, not run on hardware
//
#include <hip/hip_runtime.h>

// ---------------------------------------------------------------------------
// RGCN 3-layer decoder for MI455X (gfx1250): bf16 WMMA dense transforms with
// LDS-staged (bank-conflict-free) weight tiles, L2-resident per-relation xw
// buffer, hardware f32 atomic scatter-add, fused layer-3 edge matvec.
// ---------------------------------------------------------------------------

typedef __attribute__((ext_vector_type(8)))  __bf16 v8bf;
typedef __attribute__((ext_vector_type(16))) __bf16 v16bf;
typedef __attribute__((ext_vector_type(8)))  float  v8f;

#define RG_N_NODES 102400
#define RG_N_EDGES 1638400
#define RG_N_REL   8

// ---------------------------------------------------------------------------
// GEMM: XW = Xb[N,K](bf16, row-major) @ Wt[DOUT,K](bf16, K-contiguous = W^T)
// Block = 8 waves = 128 rows. Wt staged once per block into LDS with 16B row
// padding (row stride K+8 bf16 => lane i's ds_load_b128 starts at bank 4i:
// all 64 banks covered, zero conflicts). Each wave computes a 16-row strip x
// all DOUT columns with v_wmma_f32_16x16x32_bf16.
// ---------------------------------------------------------------------------
template <int K, int DOUT>
__global__ __launch_bounds__(256)
void rgcn_gemm_bf16(const __bf16* __restrict__ Xb,
                    const __bf16* __restrict__ Wt,
                    float* __restrict__ XW) {
    constexpr int SK = K + 8;                 // padded LDS row stride (bf16)
    __shared__ __bf16 sW[DOUT * SK];

    const int tid = threadIdx.x;

    // ---- cooperative stage: Wt (DOUT x K bf16) -> LDS, padded rows ----
    constexpr int CHUNKS = DOUT * (K / 8);    // 16-byte chunks
    for (int c = tid; c < CHUNKS; c += 256) {
        const int n  = c / (K / 8);
        const int k8 = c - n * (K / 8);
        *(v8bf*)(&sW[n * SK + k8 * 8]) =
            *(const v8bf*)(Wt + (size_t)n * K + k8 * 8);
    }
    __syncthreads();

    const int lane = tid & 31;
    const int wave = tid >> 5;
    const int half = lane >> 4;       // which 16-lane half of the wave
    const int m    = lane & 15;       // row (A) / column (B,C) index in tile
    const int rowBase = blockIdx.x * 128 + wave * 16;

    constexpr int NT = DOUT / 16;
    v8f acc[NT];
#pragma unroll
    for (int t = 0; t < NT; ++t)
#pragma unroll
        for (int i = 0; i < 8; ++i) acc[t][i] = 0.0f;

    const __bf16* arow = Xb + (size_t)(rowBase + m) * K;

#pragma unroll
    for (int k0 = 0; k0 < K; k0 += 32) {
        // A 16x32 bf16 fragment: half 0 -> K [0..7]+[16..23], half 1 -> +8
        v8bf alo = *(const v8bf*)(arow + k0 + half * 8);
        v8bf ahi = *(const v8bf*)(arow + k0 + 16 + half * 8);
        v16bf a = __builtin_shufflevector(alo, ahi,
                   0,1,2,3,4,5,6,7,8,9,10,11,12,13,14,15);
#pragma unroll
        for (int t = 0; t < NT; ++t) {
            // B 32x16 bf16 fragment: half 0 -> K [0..7]+[16..23], half 1 -> +8
            // of column m; sW rows are K-contiguous (padded stride SK).
            const __bf16* bcol = &sW[(t * 16 + m) * SK + k0 + half * 8];
            v8bf blo = *(const v8bf*)(bcol);
            v8bf bhi = *(const v8bf*)(bcol + 16);
            v16bf b = __builtin_shufflevector(blo, bhi,
                       0,1,2,3,4,5,6,7,8,9,10,11,12,13,14,15);
            acc[t] = __builtin_amdgcn_wmma_f32_16x16x32_bf16(
                         false, a, false, b, (short)0, acc[t], false, false);
        }
    }

    // C/D layout: VGPR v -> row = rowBase + v + 8*half, col = lane&15
    const int rout = rowBase + half * 8;
#pragma unroll
    for (int t = 0; t < NT; ++t) {
        float* o = XW + (size_t)rout * DOUT + t * 16 + m;
#pragma unroll
        for (int v = 0; v < 8; ++v)
            o[(size_t)v * DOUT] = acc[t][v];
    }
}

// ---------------------------------------------------------------------------
// Scatter: one wave per edge; edges of relation `rel` add ew * XW[src,:] into
// acc[dst,:] with hardware f32 atomics.
// ---------------------------------------------------------------------------
template <int DOUT>
__global__ __launch_bounds__(256)
void rgcn_scatter(const float* __restrict__ XW,
                  const int* __restrict__ ei,       // [2, E] (src row, dst row)
                  const int* __restrict__ et,       // [E]
                  const float* __restrict__ ew,     // [E]
                  int rel,
                  float* __restrict__ acc) {
    const int e    = (int)((blockIdx.x * blockDim.x + threadIdx.x) >> 5);
    const int lane = threadIdx.x & 31;
    if (e >= RG_N_EDGES) return;
    if (et[e] != rel) return;
    const int   src = ei[e];
    const int   dst = ei[RG_N_EDGES + e];
    const float w   = ew[e];
    const float* srow = XW  + (size_t)src * DOUT;
    float*       drow = acc + (size_t)dst * DOUT;
#pragma unroll
    for (int c = lane; c < DOUT; c += 32)
        unsafeAtomicAdd(drow + c, w * srow[c]);
}

// ---------------------------------------------------------------------------
// Layer 3 fused: per-edge matvec (d_out = 1) + scalar atomic scatter.
// ---------------------------------------------------------------------------
__global__ __launch_bounds__(256)
void rgcn_edge3(const float* __restrict__ h2,     // [N, 64] f32
                const int* __restrict__ ei,
                const int* __restrict__ et,
                const float* __restrict__ ew,
                const float* __restrict__ W3,     // [8, 64, 1] f32
                float* __restrict__ out) {
    const int e    = (int)((blockIdx.x * blockDim.x + threadIdx.x) >> 5);
    const int lane = threadIdx.x & 31;
    if (e >= RG_N_EDGES) return;
    const int r   = et[e];
    const int src = ei[e];
    const float* h  = h2 + (size_t)src * 64;
    const float* w3 = W3 + (size_t)r * 64;
    float p = h[lane] * w3[lane] + h[lane + 32] * w3[lane + 32];
#pragma unroll
    for (int off = 16; off > 0; off >>= 1)
        p += __shfl_down(p, off, 32);
    if (lane == 0)
        unsafeAtomicAdd(out + ei[RG_N_EDGES + e], ew[e] * p);
}

// ---------------------------------------------------------------------------
// Helpers: bias+ReLU (to bf16 for next WMMA layer, or f32), W transpose+cvt,
// fills, X conversion.
// ---------------------------------------------------------------------------
template <int DOUT>
__global__ void rgcn_bias_relu_bf16(const float* __restrict__ acc,
                                    const float* __restrict__ bias,
                                    __bf16* __restrict__ out) {
    size_t i = (size_t)blockIdx.x * blockDim.x + threadIdx.x;
    if (i >= (size_t)RG_N_NODES * DOUT) return;
    float v = acc[i] + bias[i % DOUT];
    out[i] = (__bf16)(v > 0.0f ? v : 0.0f);
}

template <int DOUT>
__global__ void rgcn_bias_relu_f32(const float* __restrict__ acc,
                                   const float* __restrict__ bias,
                                   float* __restrict__ out) {
    size_t i = (size_t)blockIdx.x * blockDim.x + threadIdx.x;
    if (i >= (size_t)RG_N_NODES * DOUT) return;
    float v = acc[i] + bias[i % DOUT];
    out[i] = v > 0.0f ? v : 0.0f;
}

// W[r][k][n] (f32) -> Wt[r][n][k] (bf16), K-contiguous for B fragments.
__global__ void rgcn_prep_w(const float* __restrict__ W,
                            __bf16* __restrict__ Wt, int K, int DOUT) {
    int i = blockIdx.x * blockDim.x + threadIdx.x;
    int total = RG_N_REL * K * DOUT;
    if (i >= total) return;
    int r   = i / (K * DOUT);
    int rem = i - r * (K * DOUT);
    int n   = rem / K;
    int k   = rem - n * K;
    Wt[(size_t)r * K * DOUT + (size_t)n * K + k] =
        (__bf16)W[((size_t)r * K + k) * DOUT + n];
}

__global__ void rgcn_f32_to_bf16(const float* __restrict__ x,
                                 __bf16* __restrict__ y, size_t n) {
    size_t i = (size_t)blockIdx.x * blockDim.x + threadIdx.x;
    if (i < n) y[i] = (__bf16)x[i];
}

__global__ void rgcn_zero_f32(float* __restrict__ p, size_t n) {
    size_t i = (size_t)blockIdx.x * blockDim.x + threadIdx.x;
    if (i < n) p[i] = 0.0f;
}

__global__ void rgcn_init_out(float* __restrict__ out,
                              const float* __restrict__ bias3) {
    size_t i = (size_t)blockIdx.x * blockDim.x + threadIdx.x;
    if (i < (size_t)RG_N_NODES) out[i] = bias3[0];
}

// ---------------------------------------------------------------------------
// Launch sequence.
// ---------------------------------------------------------------------------
extern "C" void kernel_launch(void* const* d_in, const int* in_sizes, int n_in,
                              void* d_out, int out_size, void* d_ws, size_t ws_size,
                              hipStream_t stream) {
    const float* b_z   = (const float*)d_in[0];
    const int*   ei    = (const int*)  d_in[1];   // [2, E]
    const float* ew    = (const float*)d_in[2];   // [E]
    const int*   et    = (const int*)  d_in[3];   // [E]
    const float* W1    = (const float*)d_in[4];   // [8,256,128]
    const float* bias1 = (const float*)d_in[5];
    const float* W2    = (const float*)d_in[6];   // [8,128,64]
    const float* bias2 = (const float*)d_in[7];
    const float* W3    = (const float*)d_in[8];   // [8,64,1]
    const float* bias3 = (const float*)d_in[9];
    float* out = (float*)d_out;                   // [102400]

    // Workspace layout (all 16B aligned):
    //   Xb  : bf16 node features, up to [N,256]   -> 52,428,800 B
    //   Wt  : bf16 transposed weights, <=8*256*128 -> 524,288 B
    //   XW  : f32 per-relation transform [N,128]   -> 52,428,800 B
    //   acc : f32 scatter accumulator  [N,128]     -> 52,428,800 B
    char* ws = (char*)d_ws;
    __bf16* Xb  = (__bf16*)(ws);
    __bf16* Wt  = (__bf16*)(ws + 52428800);
    float*  XW  = (float*)(ws + 52428800 + 524288);
    float*  acc = (float*)(ws + 52428800 + 524288 + 52428800);

    const int gemmBlocks = RG_N_NODES / 128;           // 800 (128 rows/block)
    const int edgeBlocks = (RG_N_EDGES * 32) / 256;    // 1 wave per edge
    auto blocksFor = [](size_t n, int bs) { return (int)((n + bs - 1) / bs); };

    // ---------------- Layer 1: [N,256] -> [N,128] ----------------
    rgcn_f32_to_bf16<<<blocksFor((size_t)RG_N_NODES * 256, 256), 256, 0, stream>>>(
        b_z, Xb, (size_t)RG_N_NODES * 256);
    rgcn_prep_w<<<blocksFor(RG_N_REL * 256 * 128, 256), 256, 0, stream>>>(
        W1, Wt, 256, 128);
    rgcn_zero_f32<<<blocksFor((size_t)RG_N_NODES * 128, 256), 256, 0, stream>>>(
        acc, (size_t)RG_N_NODES * 128);
    for (int r = 0; r < RG_N_REL; ++r) {
        rgcn_gemm_bf16<256, 128><<<gemmBlocks, 256, 0, stream>>>(
            Xb, Wt + (size_t)r * 256 * 128, XW);
        rgcn_scatter<128><<<edgeBlocks, 256, 0, stream>>>(XW, ei, et, ew, r, acc);
    }
    rgcn_bias_relu_bf16<128><<<blocksFor((size_t)RG_N_NODES * 128, 256), 256, 0, stream>>>(
        acc, bias1, Xb);  // h1 (bf16) overwrites Xb

    // ---------------- Layer 2: [N,128] -> [N,64] ----------------
    rgcn_prep_w<<<blocksFor(RG_N_REL * 128 * 64, 256), 256, 0, stream>>>(
        W2, Wt, 128, 64);
    rgcn_zero_f32<<<blocksFor((size_t)RG_N_NODES * 64, 256), 256, 0, stream>>>(
        acc, (size_t)RG_N_NODES * 64);
    for (int r = 0; r < RG_N_REL; ++r) {
        rgcn_gemm_bf16<128, 64><<<gemmBlocks, 256, 0, stream>>>(
            Xb, Wt + (size_t)r * 128 * 64, XW);
        rgcn_scatter<64><<<edgeBlocks, 256, 0, stream>>>(XW, ei, et, ew, r, acc);
    }
    // h2 (f32) written into XW buffer (free now), feeds fused layer 3
    rgcn_bias_relu_f32<64><<<blocksFor((size_t)RG_N_NODES * 64, 256), 256, 0, stream>>>(
        acc, bias2, XW);

    // ---------------- Layer 3 (fused matvec + scatter): [N,64] -> [N,1] ----
    rgcn_init_out<<<blocksFor((size_t)RG_N_NODES, 256), 256, 0, stream>>>(out, bias3);
    rgcn_edge3<<<edgeBlocks, 256, 0, stream>>>(XW, ei, et, ew, W3, out);
}
